// StyledConvBlock_32427003085142
// MI455X (gfx1250) — matrix operand order repaired and checked by reading, not compiled, not run on hardware
//
#include <hip/hip_runtime.h>
#include <cstdint>
#include <cstddef>

typedef __bf16 bf16_t;
typedef __bf16 v16bf __attribute__((ext_vector_type(16)));
typedef float  v8f   __attribute__((ext_vector_type(8)));

#define B_      4
#define C_      64
#define H_      512
#define W_      512
#define LAT_    256
#define KTAPS   9
#define KTOT    (C_ * KTAPS)   // 576
#define NCHUNK  (KTOT / 32)    // 18
#define TILE_H  8
#define TILE_C  16             // output cols per WG
#define SROWS   (TILE_H + 2)   // 10 staged rows
#define SCOLS   (TILE_C + 2)   // 18 staged cols
#define CH_PAD  72             // padded channel stride: 144B rows, 16B aligned, odd 16B stride (9 x 16B)
#define LRELU_S 0.2f
#define GAIN_   1.41421356237f
#define EPS_    1e-5f

__device__ __forceinline__ unsigned short f2bf_bits(float f) {
  union { float f; uint32_t u; } a; a.f = f;
  uint32_t r = a.u + 0x7FFFu + ((a.u >> 16) & 1u);   // round-to-nearest-even
  return (unsigned short)(r >> 16);
}
__device__ __forceinline__ bf16_t f2bf(float f) {
  union { unsigned short s; bf16_t b; } o; o.s = f2bf_bits(f);
  return o.b;
}
__device__ __forceinline__ int reflect_i(int i, int n) {
  return i < 0 ? -i : (i >= n ? 2 * n - 2 - i : i);
}

// ---------------------------------------------------------------------------
// Kernel 1: modulation s[conv][b][c] and demodulation d[conv][b][o]
// ---------------------------------------------------------------------------
__global__ void modstats_kernel(const float* __restrict__ latent,
                                const float* __restrict__ mw0, const float* __restrict__ mb0,
                                const float* __restrict__ mw1, const float* __restrict__ mb1,
                                const float* __restrict__ w0,  const float* __restrict__ w1,
                                float* __restrict__ sOut, float* __restrict__ dOut) {
  __shared__ float sSh[2 * B_ * C_];
  int t    = threadIdx.x;        // 0..511 = (conv, b, ch)
  int conv = t >> 8;
  int b    = (t >> 6) & 3;
  int ch   = t & 63;
  const float* mw = conv ? mw1 : mw0;
  const float* mb = conv ? mb1 : mb0;
  const float escale = 0.0883883476483f;   // sqrt(2/256)
  float acc = 0.f;
  for (int j = 0; j < LAT_; ++j)
    acc += latent[b * LAT_ + j] * mw[ch * LAT_ + j];
  float s = 1.f + acc * escale + mb[ch];
  sSh[t]  = s;
  sOut[t] = s;
  __syncthreads();
  const float* Wp = conv ? w1 : w0;
  const float wscale2 = 2.f / (float)KTOT;
  float dsum = 0.f;
  for (int i = 0; i < C_; ++i) {
    float si = sSh[(conv * B_ + b) * C_ + i];
    float ws = 0.f;
    for (int tp = 0; tp < KTAPS; ++tp) {
      float w = Wp[((size_t)ch * C_ + i) * KTAPS + tp];
      ws += w * w;
    }
    dsum += si * si * ws;
  }
  dOut[t] = rsqrtf(dsum * wscale2 + EPS_);
}

// ---------------------------------------------------------------------------
// Kernel 2: fragment-ready bf16 weights  w'[o,i,tap] = d_o * s_i * Ws
// Layout: wf[b][chunk(18)][mtile(4)][lane(32)][16 bf16]  (A-matrix VGPR order)
// ---------------------------------------------------------------------------
__global__ void wfrag_kernel(const float* __restrict__ w0, const float* __restrict__ w1,
                             const float* __restrict__ sArr, const float* __restrict__ dArr,
                             bf16_t* __restrict__ wf0, bf16_t* __restrict__ wf1) {
  int g = blockIdx.x * blockDim.x + threadIdx.x;
  if (g >= 2 * B_ * NCHUNK * 4 * 32) return;
  int lane  = g & 31;
  int mtile = (g >> 5) & 3;
  int rest  = g >> 7;
  int chunk = rest % NCHUNK;  rest /= NCHUNK;
  int b     = rest & 3;
  int conv  = rest >> 2;
  const float* Wp = conv ? w1 : w0;
  bf16_t*      wf = conv ? wf1 : wf0;
  const float wscale = 0.0589255650989f;  // sqrt(2/576)
  int tap = chunk >> 1, half = chunk & 1;
  int fout   = mtile * 16 + (lane & 15);
  int laneHi = (lane >> 4) & 1;
  float dv = dArr[(conv * B_ + b) * C_ + fout];
  bf16_t* dst = wf + ((((size_t)b * NCHUNK + chunk) * 4 + mtile) * 32 + lane) * 16;
  for (int e = 0; e < 16; ++e) {
    int v = e >> 1, h = e & 1;
    int K   = ((v < 4) ? 0 : 16) + laneHi * 8 + (v & 3) * 2 + h;  // ISA A-layout
    int fin = half * 32 + K;
    float sv = sArr[(conv * B_ + b) * C_ + fin];
    float w  = Wp[((size_t)fout * C_ + fin) * KTAPS + tap];
    dst[e] = f2bf(dv * sv * w * wscale);
  }
}

// ---------------------------------------------------------------------------
// Shared WMMA core: 8 waves = 4 mtiles x 2 row-quads; each wave computes
// 16 fout x (4 rows x 16 cols). 72 v_wmma per wave.
// ---------------------------------------------------------------------------
union Frag32B { uint4 q[2]; v16bf v; };

__device__ __forceinline__ void mma_core(const bf16_t* __restrict__ xlds,
                                         const bf16_t* __restrict__ wbase,
                                         int lane, int mtile, int nquad, v8f acc[4]) {
  int laneLo = lane & 15;
  int laneHi = (lane >> 4) & 1;
#pragma unroll
  for (int tap = 0; tap < KTAPS; ++tap) {
    int dy = tap / 3, dx = tap % 3;
#pragma unroll
    for (int half = 0; half < 2; ++half) {
      int chunk = tap * 2 + half;
      Frag32B af;
      const uint4* ap = reinterpret_cast<const uint4*>(
          wbase + (((size_t)chunk * 4 + mtile) * 32 + lane) * 16);
      af.q[0] = ap[0];
      af.q[1] = ap[1];
      int finBase = half * 32 + laneHi * 8;    // B-matrix K layout: K = laneHi*8 + e (e<8), 16 + laneHi*8 + (e-8)
#pragma unroll
      for (int nt = 0; nt < 4; ++nt) {
        int ro = nquad * 4 + nt;               // output row within tile
        int cc = laneLo + dx;                  // staged col (col0-1 based)
        Frag32B bf;
        const uint4* bp = reinterpret_cast<const uint4*>(
            &xlds[((dy + ro) * SCOLS + cc) * CH_PAD + finBase]);
        bf.q[0] = bp[0];
        bf.q[1] = bp[2];
        acc[nt] = __builtin_amdgcn_wmma_f32_16x16x32_bf16(
            false, af.v, false, bf.v, (short)0, acc[nt], false, false);
      }
    }
  }
}

// ---------------------------------------------------------------------------
// Conv pass 1: NCHW f32 in  ->  NHWC bf16 out (VALU staging: transpose+convert)
// ---------------------------------------------------------------------------
__global__ void __launch_bounds__(256)
conv1_kernel(const float* __restrict__ in, const bf16_t* __restrict__ wfrag,
             const float* __restrict__ bias, bf16_t* __restrict__ out) {
  __shared__ __align__(16) bf16_t xlds[SROWS * SCOLS * CH_PAD];  // 25920 B

  int bid = blockIdx.x;
  int xt  = bid & 31;
  int yt  = (bid >> 5) & 63;
  int b   = bid >> 11;
  int y0  = yt * TILE_H;
  int col0 = xt * TILE_C;
  int tid = threadIdx.x;

  for (int i = tid; i < SROWS * C_ * SCOLS; i += 256) {
    int cc = i % SCOLS;
    int c  = (i / SCOLS) & 63;
    int r  = i / (SCOLS * C_);
    int gy = reflect_i(y0 - 1 + r, H_);
    int gx = reflect_i(col0 - 1 + cc, W_);
    float v = in[(((size_t)b * C_ + c) * H_ + gy) * W_ + gx];
    xlds[(r * SCOLS + cc) * CH_PAD + c] = f2bf(v);
  }
  __syncthreads();

  int lane = tid & 31, wave = tid >> 5;
  int mtile = wave & 3, nquad = wave >> 2;
  int laneLo = lane & 15, laneHi = (lane >> 4) & 1;

  v8f acc[4];
#pragma unroll
  for (int i = 0; i < 4; ++i) { v8f z = {}; acc[i] = z; }
  const bf16_t* wbase = wfrag + (size_t)b * NCHUNK * 4 * 32 * 16;
  mma_core(xlds, wbase, lane, mtile, nquad, acc);

  // Epilogue: bias + LReLU*sqrt2, pack 8 consecutive channels -> one b128 store
  int chBase = mtile * 16 + laneHi * 8;
#pragma unroll
  for (int nt = 0; nt < 4; ++nt) {
    int grow = y0 + nquad * 4 + nt;
    int gcol = col0 + laneLo;
    unsigned short h[8];
#pragma unroll
    for (int r = 0; r < 8; ++r) {
      float v = acc[nt][r] + bias[chBase + r];
      v = (v >= 0.f ? v : LRELU_S * v) * GAIN_;
      h[r] = f2bf_bits(v);
    }
    uint4 pk;
    pk.x = (uint32_t)h[0] | ((uint32_t)h[1] << 16);
    pk.y = (uint32_t)h[2] | ((uint32_t)h[3] << 16);
    pk.z = (uint32_t)h[4] | ((uint32_t)h[5] << 16);
    pk.w = (uint32_t)h[6] | ((uint32_t)h[7] << 16);
    *reinterpret_cast<uint4*>(
        &out[(((size_t)b * H_ + grow) * W_ + gcol) * C_ + chBase]) = pk;
  }
}

// ---------------------------------------------------------------------------
// Conv pass 2: NHWC bf16 in -> NCHW f32 out.
// Staging is a straight global->LDS copy: use CDNA5 async-to-LDS (ASYNCcnt).
// ---------------------------------------------------------------------------
__global__ void __launch_bounds__(256)
conv2_kernel(const bf16_t* __restrict__ in, const bf16_t* __restrict__ wfrag,
             const float* __restrict__ bias, float* __restrict__ out) {
  __shared__ __align__(16) bf16_t xlds[SROWS * SCOLS * CH_PAD];

  int bid = blockIdx.x;
  int xt  = bid & 31;
  int yt  = (bid >> 5) & 63;
  int b   = bid >> 11;
  int y0  = yt * TILE_H;
  int col0 = xt * TILE_C;
  int tid = threadIdx.x;

  // 10 rows x 18 cols x 64ch; copy in 16B chunks (8 bf16) of the NHWC layout.
  for (int i = tid; i < SROWS * SCOLS * 8; i += 256) {
    int g  = i & 7;               // channel group of 8
    int cc = (i >> 3) % SCOLS;
    int r  = (i >> 3) / SCOLS;
    int gy = reflect_i(y0 - 1 + r, H_);
    int gx = reflect_i(col0 - 1 + cc, W_);
    const bf16_t* src = &in[(((size_t)b * H_ + gy) * W_ + gx) * C_ + g * 8];
    uint32_t lds_off = (uint32_t)(uintptr_t)(&xlds[(r * SCOLS + cc) * CH_PAD + g * 8]);
    asm volatile("global_load_async_to_lds_b128 %0, %1, off"
                 :: "v"(lds_off), "v"(src) : "memory");
  }
  asm volatile("s_wait_asynccnt 0x0" ::: "memory");
  __syncthreads();

  int lane = tid & 31, wave = tid >> 5;
  int mtile = wave & 3, nquad = wave >> 2;
  int laneLo = lane & 15, laneHi = (lane >> 4) & 1;

  v8f acc[4];
#pragma unroll
  for (int i = 0; i < 4; ++i) { v8f z = {}; acc[i] = z; }
  const bf16_t* wbase = wfrag + (size_t)b * NCHUNK * 4 * 32 * 16;
  mma_core(xlds, wbase, lane, mtile, nquad, acc);

  // Epilogue: bias + LReLU*sqrt2 -> f32 NCHW (channel-strided b32 stores)
#pragma unroll
  for (int nt = 0; nt < 4; ++nt) {
    int grow = y0 + nquad * 4 + nt;
    int gcol = col0 + laneLo;
#pragma unroll
    for (int r = 0; r < 8; ++r) {
      int fout = mtile * 16 + laneHi * 8 + r;
      float v = acc[nt][r] + bias[fout];
      v = (v >= 0.f ? v : LRELU_S * v) * GAIN_;
      out[(((size_t)b * C_ + fout) * H_ + grow) * W_ + gcol] = v;
    }
  }
}

// ---------------------------------------------------------------------------
extern "C" void kernel_launch(void* const* d_in, const int* in_sizes, int n_in,
                              void* d_out, int out_size, void* d_ws, size_t ws_size,
                              hipStream_t stream) {
  (void)in_sizes; (void)n_in; (void)out_size; (void)ws_size;
  const float* x   = (const float*)d_in[0];
  const float* lat = (const float*)d_in[1];
  const float* w0  = (const float*)d_in[2];
  const float* b0  = (const float*)d_in[3];
  const float* mw0 = (const float*)d_in[4];
  const float* mb0 = (const float*)d_in[5];
  const float* w1  = (const float*)d_in[6];
  const float* b1  = (const float*)d_in[7];
  const float* mw1 = (const float*)d_in[8];
  const float* mb1 = (const float*)d_in[9];
  float* out = (float*)d_out;

  char* ws = (char*)d_ws;
  const size_t TMP_BYTES = (size_t)B_ * C_ * H_ * W_ * sizeof(bf16_t);          // 128 MB NHWC
  const size_t WF_BYTES  = (size_t)B_ * NCHUNK * 4 * 32 * 16 * sizeof(bf16_t);  // 288 KB
  bf16_t* tmp  = (bf16_t*)ws;
  bf16_t* wf0  = (bf16_t*)(ws + TMP_BYTES);
  bf16_t* wf1  = (bf16_t*)(ws + TMP_BYTES + WF_BYTES);
  float*  sArr = (float*)(ws + TMP_BYTES + 2 * WF_BYTES);
  float*  dArr = sArr + 2 * B_ * C_;

  modstats_kernel<<<1, 512, 0, stream>>>(lat, mw0, mb0, mw1, mb1, w0, w1, sArr, dArr);
  wfrag_kernel<<<72, 256, 0, stream>>>(w0, w1, sArr, dArr, wf0, wf1);

  int grid = B_ * (H_ / TILE_H) * (W_ / TILE_C);  // 8192 workgroups
  conv1_kernel<<<grid, 256, 0, stream>>>(x,   wf0, b0, tmp);
  conv2_kernel<<<grid, 256, 0, stream>>>(tmp, wf1, b1, out);
}
